// GATConv_66992899883204
// MI455X (gfx1250) — matrix operand order repaired and checked by reading, not compiled, not run on hardware
//
#include <hip/hip_runtime.h>

#define N_NODES 50000
#define N_EDGES 800000
#define E_TOT   (N_EDGES + N_NODES)
#define D_GAT   128
#define D_LIN   512
#define D_OUT   4
#define MAX_K   512
#define LDS_PAD 8                       // halfs; pitch K+8 -> conflict-free ds_load_b128

typedef __attribute__((ext_vector_type(16))) __bf16 v16bf;
typedef __attribute__((ext_vector_type(8)))  __bf16 v8bf;
typedef __attribute__((ext_vector_type(8)))  float  v8f;

// ---------------------------------------------------------------------------
// WMMA GEMM: C[M,N] = A[M,K](bf16) * Bt[N,K](bf16, weights pre-transposed)
//            + bias, + activation (0=none, 1=relu, 2=leaky0.2)
// Block = 8 waves = 128(M)-row strip x 64(N)-col group.
// B tile staged in LDS (shared by all 8 waves). Per K-step: issue all 8
// ds_load_b128 first (partial dscnt waits overlap WMMA), branchless A
// prefetch from global, then 4 independent v_wmma back-to-back.
// ---------------------------------------------------------------------------
__global__ __launch_bounds__(256) void k_gemm_bf16(
    const __bf16* __restrict__ A, const __bf16* __restrict__ Bt,
    const float* __restrict__ bias, float* __restrict__ Cf,
    __bf16* __restrict__ Cb, int M, int N, int K, int act)
{
  __shared__ __bf16 Bs[64 * (MAX_K + LDS_PAD)];

  const int lane  = threadIdx.x & 31;
  const int wave  = threadIdx.x >> 5;
  const int colGroups = N >> 6;                 // N/64
  const int strip = blockIdx.x / colGroups;     // 128-row strip
  const int cg    = blockIdx.x - strip * colGroups;
  const int n0    = cg * 64;
  const int pitch = K + LDS_PAD;
  const int hi    = lane >> 4;
  const int l15   = lane & 15;

  // ---- cooperative fill of B tile: Bs[n][k] = Bt[n0+n][k], 16B chunks ----
  {
    const int chunksPerCol = K >> 3;            // v8bf chunks per column
    const int totalChunks  = 64 * chunksPerCol;
    for (int c = threadIdx.x; c < totalChunks; c += 256) {
      const int n    = c / chunksPerCol;
      const int koff = (c - n * chunksPerCol) << 3;
      *(v8bf*)(Bs + n * pitch + koff) =
          *(const v8bf*)(Bt + (size_t)(n0 + n) * K + koff);
    }
  }
  __syncthreads();

  // ---- per-wave 16x64 tile ----
  int rowA = strip * 128 + wave * 16 + l15;
  if (rowA >= M) rowA = M - 1;                  // clamp loads; stores guarded
  const __bf16* arow = A + (size_t)rowA * K;

  v8f acc[4] = {};
  // prefetch first A fragment (K = 8*hi + j | 16 + 8*hi + (j-8))
  v8bf aplo = *(const v8bf*)(arow + 8 * hi);
  v8bf aphi = *(const v8bf*)(arow + 16 + 8 * hi);

  // B fragment base: lane(col=l15), half-wave K offset 8*hi (+16 for hi part); col-t stride 16*pitch
  const __bf16* bbase = Bs + l15 * pitch + 8 * hi;

  for (int kc = 0; kc < K; kc += 32) {
    v16bf af = __builtin_shufflevector(aplo, aphi,
        0,1,2,3,4,5,6,7,8,9,10,11,12,13,14,15);

    // issue all 8 B ds loads up front
    const __bf16* bp = bbase + kc;
    v8bf x0 = *(const v8bf*)(bp +  0 * pitch);
    v8bf y0 = *(const v8bf*)(bp +  0 * pitch + 16);
    v8bf x1 = *(const v8bf*)(bp + 16 * pitch);
    v8bf y1 = *(const v8bf*)(bp + 16 * pitch + 16);
    v8bf x2 = *(const v8bf*)(bp + 32 * pitch);
    v8bf y2 = *(const v8bf*)(bp + 32 * pitch + 16);
    v8bf x3 = *(const v8bf*)(bp + 48 * pitch);
    v8bf y3 = *(const v8bf*)(bp + 48 * pitch + 16);

    // branchless A prefetch for next K-step (clamped; redundant on last iter)
    const int kpre = (kc + 32 < K) ? (kc + 32) : kc;
    aplo = *(const v8bf*)(arow + kpre + 8 * hi);
    aphi = *(const v8bf*)(arow + kpre + 16 + 8 * hi);

    v16bf b0 = __builtin_shufflevector(x0, y0, 0,1,2,3,4,5,6,7,8,9,10,11,12,13,14,15);
    v16bf b1 = __builtin_shufflevector(x1, y1, 0,1,2,3,4,5,6,7,8,9,10,11,12,13,14,15);
    v16bf b2 = __builtin_shufflevector(x2, y2, 0,1,2,3,4,5,6,7,8,9,10,11,12,13,14,15);
    v16bf b3 = __builtin_shufflevector(x3, y3, 0,1,2,3,4,5,6,7,8,9,10,11,12,13,14,15);

    acc[0] = __builtin_amdgcn_wmma_f32_16x16x32_bf16(
        false, af, false, b0, (short)0, acc[0], false, false);
    acc[1] = __builtin_amdgcn_wmma_f32_16x16x32_bf16(
        false, af, false, b1, (short)0, acc[1], false, false);
    acc[2] = __builtin_amdgcn_wmma_f32_16x16x32_bf16(
        false, af, false, b2, (short)0, acc[2], false, false);
    acc[3] = __builtin_amdgcn_wmma_f32_16x16x32_bf16(
        false, af, false, b3, (short)0, acc[3], false, false);
  }

  #pragma unroll
  for (int t = 0; t < 4; ++t) {
    const int col = n0 + t * 16 + l15;
    const float bv = bias ? bias[col] : 0.0f;
    #pragma unroll
    for (int r = 0; r < 8; ++r) {
      const int row = strip * 128 + wave * 16 + 8 * hi + r;  // VGPR r -> row 8*hi+r
      if (row < M) {
        float v = acc[t][r] + bv;
        if (act == 1)      v = v > 0.0f ? v : 0.0f;
        else if (act == 2) v = v > 0.0f ? v : 0.2f * v;
        const size_t idx = (size_t)row * N + col;
        if (Cf) Cf[idx] = v;
        if (Cb) Cb[idx] = (__bf16)v;
      }
    }
  }
}

// ---------------------------------------------------------------------------
// Per-node attention projections: es = h . a_src, ed = h . a_dst  (wave/node)
// ---------------------------------------------------------------------------
__global__ __launch_bounds__(256) void k_attn(const float* __restrict__ h,
    const float* __restrict__ a_src, const float* __restrict__ a_dst,
    float* __restrict__ es, float* __restrict__ ed)
{
  const int lane = threadIdx.x & 31;
  const int node = blockIdx.x * 8 + (threadIdx.x >> 5);
  if (node >= N_NODES) return;
  const float4 hv = ((const float4*)(h + (size_t)node * D_GAT))[lane];
  const float4 as = ((const float4*)a_src)[lane];
  const float4 ad = ((const float4*)a_dst)[lane];
  float s = hv.x*as.x + hv.y*as.y + hv.z*as.z + hv.w*as.w;
  float d = hv.x*ad.x + hv.y*ad.y + hv.z*ad.z + hv.w*ad.w;
  #pragma unroll
  for (int off = 16; off > 0; off >>= 1) {
    s += __shfl_xor(s, off, 32);
    d += __shfl_xor(d, off, 32);
  }
  if (lane == 0) { es[node] = s; ed[node] = d; }
}

// order-preserving float<->uint key for exact atomic max
__device__ __forceinline__ unsigned f2key(float f) {
  int i = __float_as_int(f);
  return (i >= 0) ? ((unsigned)i ^ 0x80000000u) : ~(unsigned)i;
}
__device__ __forceinline__ float key2f(unsigned k) {
  return (k & 0x80000000u) ? __int_as_float((int)(k ^ 0x80000000u))
                           : __int_as_float((int)~k);
}

__global__ void k_init(float* __restrict__ agg, unsigned* __restrict__ mkey,
                       float* __restrict__ denom)
{
  const int tid = blockIdx.x * blockDim.x + threadIdx.x;
  if (tid < N_NODES * D_GAT) agg[tid] = 0.0f;
  if (tid < N_NODES) { mkey[tid] = 0u; denom[tid] = 0.0f; }
}

__global__ void k_edge_score(const int* __restrict__ eidx,
    const float* __restrict__ es, const float* __restrict__ ed,
    float* __restrict__ elr, unsigned* __restrict__ mkey)
{
  const int t = blockIdx.x * blockDim.x + threadIdx.x;
  if (t >= E_TOT) return;
  const int s = (t < N_EDGES) ? eidx[t]            : (t - N_EDGES);
  const int d = (t < N_EDGES) ? eidx[N_EDGES + t]  : (t - N_EDGES);
  float e = es[s] + ed[d];
  e = e > 0.0f ? e : 0.2f * e;                 // leaky_relu(e, 0.2)
  elr[t] = e;
  atomicMax(&mkey[d], f2key(e));               // exact, order-independent
}

__global__ void k_edge_exp(const int* __restrict__ eidx,
    const float* __restrict__ elr, const unsigned* __restrict__ mkey,
    float* __restrict__ enm, float* __restrict__ denom)
{
  const int t = blockIdx.x * blockDim.x + threadIdx.x;
  if (t >= E_TOT) return;
  const int d = (t < N_EDGES) ? eidx[N_EDGES + t] : (t - N_EDGES);
  const float ex = __expf(elr[t] - key2f(mkey[d]));
  enm[t] = ex;
  atomicAdd(&denom[d], ex);
}

// one wave per edge: lane covers 4 feature dims; h[src] read coalesced
__global__ __launch_bounds__(256) void k_edge_agg(const int* __restrict__ eidx,
    const float* __restrict__ enm, const float* __restrict__ denom,
    const float* __restrict__ h, float* __restrict__ agg)
{
  const int lane = threadIdx.x & 31;
  const int t = blockIdx.x * 8 + (threadIdx.x >> 5);
  if (t >= E_TOT) return;
  const int s = (t < N_EDGES) ? eidx[t]           : (t - N_EDGES);
  const int d = (t < N_EDGES) ? eidx[N_EDGES + t] : (t - N_EDGES);
  const float alpha = enm[t] / denom[d];
  const float4 hv = ((const float4*)(h + (size_t)s * D_GAT))[lane];
  float* ap = agg + (size_t)d * D_GAT + lane * 4;
  atomicAdd(ap + 0, alpha * hv.x);
  atomicAdd(ap + 1, alpha * hv.y);
  atomicAdd(ap + 2, alpha * hv.z);
  atomicAdd(ap + 3, alpha * hv.w);
}

__global__ void k_node_finish(const float* __restrict__ agg,
    const float* __restrict__ b, __bf16* __restrict__ xb)
{
  const int tid = blockIdx.x * blockDim.x + threadIdx.x;
  if (tid >= N_NODES * D_GAT) return;
  float v = agg[tid] + b[tid & (D_GAT - 1)];
  v = v > 0.0f ? v : 0.2f * v;                 // leaky_relu after conv
  xb[tid] = (__bf16)v;
}

__global__ void k_cast_bf16(const float* __restrict__ in,
                            __bf16* __restrict__ out, int n)
{
  const int tid = blockIdx.x * blockDim.x + threadIdx.x;
  if (tid < n) out[tid] = (__bf16)in[tid];
}

__global__ void k_transpose_bf16(const float* __restrict__ W,
                                 __bf16* __restrict__ Wt, int K, int N)
{
  const int tid = blockIdx.x * blockDim.x + threadIdx.x;
  if (tid >= K * N) return;
  const int k = tid / N, n = tid - k * N;
  Wt[(size_t)n * K + k] = (__bf16)W[tid];
}

// final 512 -> 4 projection (tiny; VALU dot per output element)
__global__ void k_head(const __bf16* __restrict__ y,
    const float* __restrict__ Wout, const float* __restrict__ bout,
    float* __restrict__ out)
{
  const int tid = blockIdx.x * blockDim.x + threadIdx.x;
  if (tid >= N_NODES * D_OUT) return;
  const int n = tid >> 2, o = tid & 3;
  const __bf16* yr = y + (size_t)n * D_LIN;
  float acc = bout[o];
  #pragma unroll 8
  for (int k = 0; k < D_LIN; ++k)
    acc += (float)yr[k] * Wout[k * D_OUT + o];
  out[tid] = acc;
}

// ---------------------------------------------------------------------------
extern "C" void kernel_launch(void* const* d_in, const int* in_sizes, int n_in,
                              void* d_out, int out_size, void* d_ws, size_t ws_size,
                              hipStream_t stream)
{
  const float* x        = (const float*)d_in[0];
  const int*   eidx     = (const int*)  d_in[1];
  const float* conv_W   = (const float*)d_in[2];
  const float* conv_as  = (const float*)d_in[3];
  const float* conv_ad  = (const float*)d_in[4];
  const float* conv_b   = (const float*)d_in[5];
  const float* lin_W0   = (const float*)d_in[6];
  const float* lin_b0   = (const float*)d_in[7];
  const float* lin_W1   = (const float*)d_in[8];
  const float* lin_b1   = (const float*)d_in[9];
  const float* lin_W2   = (const float*)d_in[10];
  const float* lin_b2   = (const float*)d_in[11];
  const float* lin_Wout = (const float*)d_in[12];
  const float* lin_bout = (const float*)d_in[13];
  float* out = (float*)d_out;
  (void)in_sizes; (void)n_in; (void)out_size; (void)ws_size;

  char* p = (char*)d_ws;
  auto alloc = [&](size_t bytes) -> void* {
    char* r = p; p += (bytes + 255) & ~(size_t)255; return (void*)r;
  };
  __bf16*   xb  = (__bf16*)  alloc((size_t)N_NODES * D_GAT * 2);
  float*    h   = (float*)   alloc((size_t)N_NODES * D_GAT * 4);
  float*    agg = (float*)   alloc((size_t)N_NODES * D_GAT * 4);
  float*    es  = (float*)   alloc((size_t)N_NODES * 4);
  float*    ed  = (float*)   alloc((size_t)N_NODES * 4);
  float*    elr = (float*)   alloc((size_t)E_TOT * 4);
  float*    enm = (float*)   alloc((size_t)E_TOT * 4);
  unsigned* mk  = (unsigned*)alloc((size_t)N_NODES * 4);
  float*    den = (float*)   alloc((size_t)N_NODES * 4);
  __bf16*   Wct = (__bf16*)  alloc((size_t)3 * D_GAT * D_GAT * 2);
  __bf16*   W0t = (__bf16*)  alloc((size_t)D_LIN * D_GAT * 2);
  __bf16*   W1t = (__bf16*)  alloc((size_t)D_LIN * D_LIN * 2);
  __bf16*   W2t = (__bf16*)  alloc((size_t)D_LIN * D_LIN * 2);
  __bf16*   yA  = (__bf16*)  alloc((size_t)N_NODES * D_LIN * 2);
  __bf16*   yB  = (__bf16*)  alloc((size_t)N_NODES * D_LIN * 2);

  const int T = 256;
  // --- weight prep: transpose + bf16 cast (cheap, deterministic)
  for (int i = 0; i < 3; ++i)
    k_transpose_bf16<<<(D_GAT*D_GAT + T-1)/T, T, 0, stream>>>(
        conv_W + (size_t)i*D_GAT*D_GAT, Wct + (size_t)i*D_GAT*D_GAT, D_GAT, D_GAT);
  k_transpose_bf16<<<(D_GAT*D_LIN + T-1)/T, T, 0, stream>>>(lin_W0, W0t, D_GAT, D_LIN);
  k_transpose_bf16<<<(D_LIN*D_LIN + T-1)/T, T, 0, stream>>>(lin_W1, W1t, D_LIN, D_LIN);
  k_transpose_bf16<<<(D_LIN*D_LIN + T-1)/T, T, 0, stream>>>(lin_W2, W2t, D_LIN, D_LIN);
  k_cast_bf16<<<(N_NODES*D_GAT + T-1)/T, T, 0, stream>>>(x, xb, N_NODES*D_GAT);

  const int mStrips = (N_NODES + 127) / 128;    // 128-row strips per block

  // --- 3 GATConv layers
  for (int i = 0; i < 3; ++i) {
    k_gemm_bf16<<<mStrips * (D_GAT/64), T, 0, stream>>>(
        xb, Wct + (size_t)i*D_GAT*D_GAT, nullptr, h, nullptr,
        N_NODES, D_GAT, D_GAT, 0);
    k_attn<<<(N_NODES + 7)/8, T, 0, stream>>>(
        h, conv_as + i*D_GAT, conv_ad + i*D_GAT, es, ed);
    k_init<<<(N_NODES*D_GAT + T-1)/T, T, 0, stream>>>(agg, mk, den);
    k_edge_score<<<(E_TOT + T-1)/T, T, 0, stream>>>(eidx, es, ed, elr, mk);
    k_edge_exp<<<(E_TOT + T-1)/T, T, 0, stream>>>(eidx, elr, mk, enm, den);
    k_edge_agg<<<(E_TOT + 7)/8, T, 0, stream>>>(eidx, enm, den, h, agg);
    k_node_finish<<<(N_NODES*D_GAT + T-1)/T, T, 0, stream>>>(
        agg, conv_b + i*D_GAT, xb);
  }

  // --- MLP head (WMMA GEMMs, relu fused, bf16 ping-pong)
  k_gemm_bf16<<<mStrips * (D_LIN/64), T, 0, stream>>>(
      xb, W0t, lin_b0, nullptr, yA, N_NODES, D_LIN, D_GAT, 1);
  k_gemm_bf16<<<mStrips * (D_LIN/64), T, 0, stream>>>(
      yA, W1t, lin_b1, nullptr, yB, N_NODES, D_LIN, D_LIN, 1);
  k_gemm_bf16<<<mStrips * (D_LIN/64), T, 0, stream>>>(
      yB, W2t, lin_b2, nullptr, yA, N_NODES, D_LIN, D_LIN, 1);

  k_head<<<(N_NODES*D_OUT + T-1)/T, T, 0, stream>>>(yA, lin_Wout, lin_bout, out);
}
